// FourOverSixQwenExperts_10290741641386
// MI455X (gfx1250) — matrix operand order repaired and checked by reading, not compiled, not run on hardware
//
#include <hip/hip_runtime.h>
#include <hip/hip_bf16.h>
#include <cstdint>

typedef __attribute__((ext_vector_type(16))) _Float16 v16h;
typedef __attribute__((ext_vector_type(8)))  float    v8f;

#define NEXPERT 32
#define HIDDEN  1024
#define INTER   512
#define TOKENS  4096
#define TOPK    4
#define NPAIR   (TOKENS*TOPK)   // 16384
#define CAP     1024
#define QB      16

// ---------------- workspace layout (bytes) ----------------
static const size_t OFF_XQ    = 0;                                              // f16 [TOKENS][HIDDEN]
static const size_t OFF_GUT   = OFF_XQ    + (size_t)TOKENS*HIDDEN*2;            // f16 [E][2I][H] (transposed)
static const size_t OFF_DNT   = OFF_GUT   + (size_t)NEXPERT*2*INTER*HIDDEN*2;   // f16 [E][H][I]  (transposed)
static const size_t OFF_CNT   = OFF_DNT   + (size_t)NEXPERT*INTER*HIDDEN*2;     // int [E]
static const size_t OFF_SPID  = OFF_CNT   + 256;                                // int [E*CAP] pair-id per slot
static const size_t OFF_PLOC  = OFF_SPID  + (size_t)NEXPERT*CAP*4;              // int [NPAIR] slot or -1
static const size_t OFF_GATED = OFF_PLOC  + (size_t)NPAIR*4;                    // f16 [E*CAP][I]
static const size_t OFF_OUTP  = OFF_GATED + (size_t)NEXPERT*CAP*INTER*2;        // f16 [NPAIR][H]

// ---------------- e2m1 fake quant helper ----------------
__device__ __forceinline__ float e2m1_round(float y) {
    float a = fabsf(y);
    float g;
    if      (a <= 0.25f) g = 0.0f;
    else if (a <= 0.75f) g = 0.5f;
    else if (a <= 1.25f) g = 1.0f;
    else if (a <= 1.75f) g = 1.5f;
    else if (a <= 2.5f ) g = 2.0f;
    else if (a <= 3.5f ) g = 3.0f;
    else if (a <= 5.0f ) g = 4.0f;
    else                 g = 6.0f;
    return copysignf(g, y);
}

// LDS byte offset of a shared-memory pointer: generic LDS addresses carry the
// DS offset in the low 32 bits.
__device__ __forceinline__ unsigned lds_off(const void* p) {
    return (unsigned)(uintptr_t)p;
}
__device__ __forceinline__ void async_copy_b128(unsigned loff, const void* gaddr) {
    asm volatile("global_load_async_to_lds_b128 %0, %1, off" :: "v"(loff), "v"(gaddr) : "memory");
}
__device__ __forceinline__ void wait_async0() {
    asm volatile("s_wait_asynccnt 0x0" ::: "memory");
}

// ---------------- dispatch: assign capacity slots per expert ----------------
__global__ void moe_dispatch_kernel(const int* __restrict__ topk_idx,
                                    int* __restrict__ counts,
                                    int* __restrict__ spid,
                                    int* __restrict__ ploc) {
    int t = blockIdx.x * blockDim.x + threadIdx.x;
    if (t >= NPAIR) return;
    int token = t >> 2;
    int k     = t & 3;
    int e     = topk_idx[token * TOPK + k];
    int slot  = atomicAdd(&counts[e], 1);
    if (slot < CAP) {
        spid[e * CAP + slot] = t;
        ploc[t] = e * CAP + slot;
    } else {
        ploc[t] = -1;
    }
}

// ---------------- activation qdq: fp32 -> e2m1-qdq -> f16 ----------------
__global__ void quant_x_kernel(const float* __restrict__ hs, _Float16* __restrict__ xq) {
    int tid = blockIdx.x * blockDim.x + threadIdx.x;       // TOKENS * (HIDDEN/QB)
    if (tid >= TOKENS * (HIDDEN / QB)) return;
    int b     = tid % (HIDDEN / QB);
    int token = tid / (HIDDEN / QB);
    const float* src = hs + (size_t)token * HIDDEN + b * QB;
    float v[QB];
    float am = 0.0f;
    #pragma unroll
    for (int j = 0; j < QB; ++j) { v[j] = src[j]; am = fmaxf(am, fabsf(v[j])); }
    float s = am * (1.0f / 6.0f);
    s = (s > 0.0f) ? s : 1.0f;
    __align__(16) _Float16 q[QB];
    #pragma unroll
    for (int j = 0; j < QB; ++j) q[j] = (_Float16)(e2m1_round(v[j] / s) * s);
    _Float16* dst = xq + (size_t)token * HIDDEN + b * QB;
    *(uint4*)(dst)     = *(uint4*)(q);
    *(uint4*)(dst + 8) = *(uint4*)(q + 8);
}

// ---------------- gate_up qdq + transpose: [E][H][2I] -> f16 [E][2I][H] ----------------
__global__ void quant_gu_kernel(const float* __restrict__ gu, _Float16* __restrict__ guT) {
    int tid = blockIdx.x * blockDim.x + threadIdx.x;       // E * (2I/QB) * H
    if (tid >= NEXPERT * (2 * INTER / QB) * HIDDEN) return;
    int h  = tid % HIDDEN;
    int t2 = tid / HIDDEN;
    int b  = t2 % (2 * INTER / QB);
    int e  = t2 / (2 * INTER / QB);
    const float* src = gu + ((size_t)e * HIDDEN + h) * (2 * INTER) + b * QB;
    float v[QB];
    float am = 0.0f;
    #pragma unroll
    for (int j = 0; j < QB; ++j) { v[j] = src[j]; am = fmaxf(am, fabsf(v[j])); }
    float s = am * (1.0f / 6.0f);
    s = (s > 0.0f) ? s : 1.0f;
    #pragma unroll
    for (int j = 0; j < QB; ++j)
        guT[((size_t)e * 2 * INTER + b * QB + j) * HIDDEN + h] = (_Float16)(e2m1_round(v[j] / s) * s);
}

// ---------------- down qdq + transpose: [E][I][H] -> f16 [E][H][I] ----------------
__global__ void quant_dn_kernel(const float* __restrict__ dn, _Float16* __restrict__ dnT) {
    int tid = blockIdx.x * blockDim.x + threadIdx.x;       // E * (H/QB) * I
    if (tid >= NEXPERT * (HIDDEN / QB) * INTER) return;
    int i  = tid % INTER;
    int t2 = tid / INTER;
    int b  = t2 % (HIDDEN / QB);
    int e  = t2 / (HIDDEN / QB);
    const float* src = dn + ((size_t)e * INTER + i) * HIDDEN + b * QB;
    float v[QB];
    float am = 0.0f;
    #pragma unroll
    for (int j = 0; j < QB; ++j) { v[j] = src[j]; am = fmaxf(am, fabsf(v[j])); }
    float s = am * (1.0f / 6.0f);
    s = (s > 0.0f) ? s : 1.0f;
    #pragma unroll
    for (int j = 0; j < QB; ++j)
        dnT[((size_t)e * HIDDEN + b * QB + j) * INTER + i] = (_Float16)(e2m1_round(v[j] / s) * s);
}

// ---------------- GEMM1: A panel staged in LDS (async), 4 waves x 8 col-tiles ----------------
// LDS row stride padded to 1032 halves (2064 B == 16 mod 256) -> ds_load_b128 conflict-free fan-out.
#define A1_STRIDE 1032
__global__ void __launch_bounds__(128)
gemm1_kernel(const _Float16* __restrict__ xq,
             const _Float16* __restrict__ guT,
             const int* __restrict__ counts,
             const int* __restrict__ spid,
             _Float16* __restrict__ gated) {
    __shared__ _Float16 sA[16 * A1_STRIDE];
    const int rt = blockIdx.x;          // CAP/16 row tiles
    const int e  = blockIdx.y;
    const int count = counts[e];
    if (rt * 16 >= count) return;       // uniform across block

    const int tid = threadIdx.x;
    // stage A: 16 rows x 1024 halves; one async b128 per thread per row
    for (int r = 0; r < 16; ++r) {
        int pid   = spid[e * CAP + rt * 16 + r];   // uniform per r
        int token = pid >> 2;
        const _Float16* g = xq + (size_t)token * HIDDEN + tid * 8;
        async_copy_b128(lds_off(&sA[r * A1_STRIDE + tid * 8]), g);
    }
    wait_async0();
    __syncthreads();

    const int lane = tid & 31;
    const int w    = tid >> 5;          // wave id 0..3
    const int half = lane >> 4;
    const int l16  = lane & 15;
    const int aoff = half * 16;         // A: K {0..15} / {16..31}
    const int boff = half * 16;         // B: K {0..15} / {16..31}

    v8f accg[8] = {};
    v8f accu[8] = {};

    for (int kb = 0; kb < HIDDEN; kb += 32) {
        union { v16h v; uint4 q[2]; } A;
        A.q[0] = *(const uint4*)&sA[l16 * A1_STRIDE + kb + aoff];
        A.q[1] = *(const uint4*)&sA[l16 * A1_STRIDE + kb + aoff + 8];
        #pragma unroll
        for (int j = 0; j < 8; ++j) {
            const int ct = j * 4 + w;
            const _Float16* bg = guT + ((size_t)e * 2 * INTER + ct * 16 + l16) * HIDDEN;
            const _Float16* bu = bg + (size_t)INTER * HIDDEN;
            union { v16h v; uint4 q[2]; } Bg, Bu;
            Bg.q[0] = *(const uint4*)(bg + kb + boff);
            Bg.q[1] = *(const uint4*)(bg + kb + boff + 8);
            Bu.q[0] = *(const uint4*)(bu + kb + boff);
            Bu.q[1] = *(const uint4*)(bu + kb + boff + 8);
            if (j == 0) {
                __builtin_prefetch(bg + kb + 32 + boff, 0, 1);
                __builtin_prefetch(bu + kb + 32 + boff, 0, 1);
            }
            accg[j] = __builtin_amdgcn_wmma_f32_16x16x32_f16(false, A.v, false, Bg.v, (short)0, accg[j], false, false);
            accu[j] = __builtin_amdgcn_wmma_f32_16x16x32_f16(false, A.v, false, Bu.v, (short)0, accu[j], false, false);
        }
    }

    // epilogue: silu(gate)*up, block-16 qdq along INTER (16 lanes of each half = one block)
    #pragma unroll
    for (int j = 0; j < 8; ++j) {
        const int ct = j * 4 + w;
        #pragma unroll
        for (int r = 0; r < 8; ++r) {
            float g = accg[j][r];
            float u = accu[j][r];
            float sig = __builtin_amdgcn_rcpf(1.0f + __expf(-g));   // v_rcp_f32, no IEEE divide
            float act = g * sig * u;
            float am = fabsf(act);
            am = fmaxf(am, __shfl_xor(am, 1, 32));
            am = fmaxf(am, __shfl_xor(am, 2, 32));
            am = fmaxf(am, __shfl_xor(am, 4, 32));
            am = fmaxf(am, __shfl_xor(am, 8, 32));
            float s = am * (1.0f / 6.0f);
            s = (s > 0.0f) ? s : 1.0f;
            float q = e2m1_round(act * __builtin_amdgcn_rcpf(s)) * s;
            int rowl = r + half * 8;
            gated[((size_t)e * CAP + rt * 16 + rowl) * INTER + ct * 16 + l16] = (_Float16)q;
        }
    }
}

// ---------------- GEMM2: A panel staged in LDS (async), 4 waves x 16 col-tiles ----------------
#define A2_STRIDE 520   // 512 + 8 halves padding (1040 B == 16 mod 256)
__global__ void __launch_bounds__(128)
gemm2_kernel(const _Float16* __restrict__ gated,
             const _Float16* __restrict__ dnT,
             const int* __restrict__ counts,
             const int* __restrict__ spid,
             _Float16* __restrict__ outp) {
    __shared__ _Float16 sA[16 * A2_STRIDE];
    const int rt = blockIdx.x;
    const int e  = blockIdx.y;
    const int count = counts[e];
    if (rt * 16 >= count) return;

    const int tid = threadIdx.x;
    const _Float16* abase = gated + ((size_t)e * CAP + rt * 16) * INTER;
    // stage A: 16 rows x 512 halves = 1024 b128 chunks; 128 threads x 8 iters
    #pragma unroll
    for (int it = 0; it < 8; ++it) {
        int c    = it * 128 + tid;
        int r    = c >> 6;          // 64 chunks per row
        int col8 = c & 63;
        async_copy_b128(lds_off(&sA[r * A2_STRIDE + col8 * 8]),
                        abase + (size_t)r * INTER + col8 * 8);
    }
    wait_async0();
    __syncthreads();

    const int lane = tid & 31;
    const int w    = tid >> 5;
    const int half = lane >> 4;
    const int l16  = lane & 15;
    const int aoff = half * 16;
    const int boff = half * 16;

    v8f acc[16] = {};

    for (int kb = 0; kb < INTER; kb += 32) {
        union { v16h v; uint4 q[2]; } A;
        A.q[0] = *(const uint4*)&sA[l16 * A2_STRIDE + kb + aoff];
        A.q[1] = *(const uint4*)&sA[l16 * A2_STRIDE + kb + aoff + 8];
        #pragma unroll
        for (int j = 0; j < 16; ++j) {
            const int ct = j * 4 + w;
            const _Float16* bptr = dnT + ((size_t)e * HIDDEN + ct * 16 + l16) * INTER;
            union { v16h v; uint4 q[2]; } B;
            B.q[0] = *(const uint4*)(bptr + kb + boff);
            B.q[1] = *(const uint4*)(bptr + kb + boff + 8);
            if (j == 0) __builtin_prefetch(bptr + kb + 32 + boff, 0, 1);
            acc[j] = __builtin_amdgcn_wmma_f32_16x16x32_f16(false, A.v, false, B.v, (short)0, acc[j], false, false);
        }
    }

    #pragma unroll
    for (int j = 0; j < 16; ++j) {
        const int ct = j * 4 + w;
        #pragma unroll
        for (int r = 0; r < 8; ++r) {
            int rowl = r + half * 8;
            int slot = rt * 16 + rowl;
            if (slot < count) {
                int pid = spid[e * CAP + slot];
                outp[(size_t)pid * HIDDEN + ct * 16 + l16] = (_Float16)acc[j][r];
            }
        }
    }
}

// ---------------- combine: deterministic weighted sum over top-k ----------------
__global__ void combine_kernel(const _Float16* __restrict__ outp,
                               const float* __restrict__ tkw,
                               const int* __restrict__ ploc,
                               float* __restrict__ out) {
    int tid = blockIdx.x * blockDim.x + threadIdx.x;
    if (tid >= TOKENS * HIDDEN) return;
    int h     = tid & (HIDDEN - 1);
    int token = tid >> 10;
    float s = 0.0f;
    #pragma unroll
    for (int k = 0; k < TOPK; ++k) {
        int t = token * TOPK + k;
        if (ploc[t] >= 0)
            s += tkw[t] * (float)outp[(size_t)t * HIDDEN + h];
    }
    out[tid] = s;
}

// ---------------- host launcher ----------------
extern "C" void kernel_launch(void* const* d_in, const int* in_sizes, int n_in,
                              void* d_out, int out_size, void* d_ws, size_t ws_size,
                              hipStream_t stream) {
    const float* hidden_states = (const float*)d_in[0];
    const int*   topk_idx      = (const int*)  d_in[1];
    const float* topk_w        = (const float*)d_in[2];
    const float* gate_up       = (const float*)d_in[3];
    const float* down          = (const float*)d_in[4];
    float* out = (float*)d_out;

    char* ws = (char*)d_ws;
    _Float16* xq    = (_Float16*)(ws + OFF_XQ);
    _Float16* guT   = (_Float16*)(ws + OFF_GUT);
    _Float16* dnT   = (_Float16*)(ws + OFF_DNT);
    int*      cnt   = (int*)     (ws + OFF_CNT);
    int*      spid  = (int*)     (ws + OFF_SPID);
    int*      ploc  = (int*)     (ws + OFF_PLOC);
    _Float16* gated = (_Float16*)(ws + OFF_GATED);
    _Float16* outp  = (_Float16*)(ws + OFF_OUTP);

    hipMemsetAsync(cnt,  0, 256, stream);
    hipMemsetAsync(spid, 0, (size_t)NEXPERT * CAP * 4, stream);

    moe_dispatch_kernel<<<(NPAIR + 255) / 256, 256, 0, stream>>>(topk_idx, cnt, spid, ploc);

    quant_x_kernel <<<(TOKENS * (HIDDEN / QB) + 255) / 256, 256, 0, stream>>>(hidden_states, xq);
    quant_gu_kernel<<<(NEXPERT * (2 * INTER / QB) * HIDDEN + 255) / 256, 256, 0, stream>>>(gate_up, guT);
    quant_dn_kernel<<<(NEXPERT * (HIDDEN / QB) * INTER + 255) / 256, 256, 0, stream>>>(down, dnT);

    gemm1_kernel<<<dim3(CAP / 16, NEXPERT), 128, 0, stream>>>(xq, guT, cnt, spid, gated);
    gemm2_kernel<<<dim3(CAP / 16, NEXPERT), 128, 0, stream>>>(gated, dnT, cnt, spid, outp);

    combine_kernel<<<(TOKENS * HIDDEN + 255) / 256, 256, 0, stream>>>(outp, topk_w, ploc, out);
}
